// InvariantRegionNetwork_35347580846893
// MI455X (gfx1250) — compile-verified, matrix operand order checked by reading, not run on hardware
//
#include <hip/hip_runtime.h>
#include <hip/hip_bf16.h>

// ---------------------------------------------------------------------------
// CDNA5 (gfx1250) wave32 WMMA types
// ---------------------------------------------------------------------------
typedef __attribute__((ext_vector_type(16))) __bf16 v16bf;
typedef __attribute__((ext_vector_type(8)))  float  v8f;
typedef __attribute__((ext_vector_type(4)))  float  f32x4;

union FragU {
    v16bf bv;
    f32x4 f4[2];
};

// ---------------------------------------------------------------------------
// Weight convert + transpose: W fp32 [Kd, Nc] -> Wt bf16 [Nc, Kd]
// ---------------------------------------------------------------------------
__global__ __launch_bounds__(256) void irn_convw(
    const float* __restrict__ W, __hip_bfloat16* __restrict__ Wt, int Kd, int Nc)
{
    int t = blockIdx.x * 256 + threadIdx.x;
    if (t >= Kd * Nc) return;
    int n = t / Kd;
    int k = t - n * Kd;
    Wt[t] = __float2bfloat16(W[(size_t)k * Nc + n]);
}

// ---------------------------------------------------------------------------
// LayerNorm over D=256, one wave per row (8 elems/lane), bf16 output
// ---------------------------------------------------------------------------
__global__ __launch_bounds__(256) void irn_ln256(
    const float* __restrict__ x, const float* __restrict__ g, const float* __restrict__ b,
    __hip_bfloat16* __restrict__ out, int N)
{
    int wave = threadIdx.x >> 5, lane = threadIdx.x & 31;
    int row = blockIdx.x * 8 + wave;
    if (row >= N) return;
    const float* xr = x + (size_t)row * 256;
    float v[8], s = 0.f, ss = 0.f;
#pragma unroll
    for (int i = 0; i < 8; ++i) { v[i] = xr[lane + 32 * i]; s += v[i]; ss += v[i] * v[i]; }
#pragma unroll
    for (int o = 16; o >= 1; o >>= 1) { s += __shfl_xor(s, o, 32); ss += __shfl_xor(ss, o, 32); }
    float m  = s * (1.f / 256.f);
    float va = ss * (1.f / 256.f) - m * m;
    float rs = rsqrtf(va + 1e-5f);
#pragma unroll
    for (int i = 0; i < 8; ++i) {
        int c = lane + 32 * i;
        out[(size_t)row * 256 + c] = __float2bfloat16((v[i] - m) * rs * g[c] + b[c]);
    }
}

// ---------------------------------------------------------------------------
// WMMA GEMM, 2x2 register-blocked: each wave computes a 32x32 output tile
// (4 x v_wmma_f32_16x16x32_bf16 per k-step from 8 b128 fragment loads).
//   A bf16 [M,Kd] row-major; Wt bf16 [Nc,Kd] row-major (pre-transposed), so
//   A and B fragments are identical contiguous 16-byte loads per the CDNA5
//   16-bit A-layout:
//     lane<16 : row tm+lane,    K = k0+{0..7} and k0+{16..23}
//     lane>=16: row tm+lane-16, K = k0+{8..15} and k0+{24..31}
//   C/D layout: VGPR r -> M = tm + (lane<16?0:8) + r, N = tn + (lane&15)
// Halves L2 re-reads of both activations and weights vs 1-tile waves.
// ---------------------------------------------------------------------------
__global__ __launch_bounds__(256) void irn_gemm_bf16(
    const __hip_bfloat16* __restrict__ A,
    const __hip_bfloat16* __restrict__ Wt,
    const float* __restrict__ bias,
    const float* __restrict__ resid,
    float* __restrict__ outF,
    __hip_bfloat16* __restrict__ outB,
    int M, int Kd, int Nc, int relu)
{
    int wave = threadIdx.x >> 5;
    int lane = threadIdx.x & 31;
    int tilesN = Nc >> 5;                 // 32-wide macro tiles
    int tile = blockIdx.x * 8 + wave;
    int totalTiles = (M >> 5) * tilesN;
    if (tile >= totalTiles) return;
    int tm = (tile / tilesN) << 5;
    int tn = (tile % tilesN) << 5;

    int l16 = lane & 15;
    int hi  = lane >> 4; // 0 or 1
    const __hip_bfloat16* a0 = A  + (size_t)(tm + l16) * Kd + hi * 8;
    const __hip_bfloat16* a1 = a0 + (size_t)16 * Kd;
    const __hip_bfloat16* b0 = Wt + (size_t)(tn + l16) * Kd + hi * 8;
    const __hip_bfloat16* b1 = b0 + (size_t)16 * Kd;

    v8f c00 = {}, c01 = {}, c10 = {}, c11 = {};
    for (int k0 = 0; k0 < Kd; k0 += 32) {
        FragU fa0, fa1, fb0, fb1;
        fa0.f4[0] = *reinterpret_cast<const f32x4*>(a0 + k0);
        fa0.f4[1] = *reinterpret_cast<const f32x4*>(a0 + k0 + 16);
        fa1.f4[0] = *reinterpret_cast<const f32x4*>(a1 + k0);
        fa1.f4[1] = *reinterpret_cast<const f32x4*>(a1 + k0 + 16);
        fb0.f4[0] = *reinterpret_cast<const f32x4*>(b0 + k0);
        fb0.f4[1] = *reinterpret_cast<const f32x4*>(b0 + k0 + 16);
        fb1.f4[0] = *reinterpret_cast<const f32x4*>(b1 + k0);
        fb1.f4[1] = *reinterpret_cast<const f32x4*>(b1 + k0 + 16);
        // gfx1250 global_prefetch_b8 for the next K-chunk
        __builtin_prefetch(a0 + k0 + 64, 0, 3);
        __builtin_prefetch(a1 + k0 + 64, 0, 3);
        __builtin_prefetch(b0 + k0 + 64, 0, 3);
        __builtin_prefetch(b1 + k0 + 64, 0, 3);
        c00 = __builtin_amdgcn_wmma_f32_16x16x32_bf16(false, fa0.bv, false, fb0.bv,
                                                      (short)0, c00, false, false);
        c01 = __builtin_amdgcn_wmma_f32_16x16x32_bf16(false, fa0.bv, false, fb1.bv,
                                                      (short)0, c01, false, false);
        c10 = __builtin_amdgcn_wmma_f32_16x16x32_bf16(false, fa1.bv, false, fb0.bv,
                                                      (short)0, c10, false, false);
        c11 = __builtin_amdgcn_wmma_f32_16x16x32_bf16(false, fa1.bv, false, fb1.bv,
                                                      (short)0, c11, false, false);
    }

    v8f accs[4] = { c00, c01, c10, c11 };
#pragma unroll
    for (int mi = 0; mi < 2; ++mi) {
#pragma unroll
        for (int ni = 0; ni < 2; ++ni) {
            v8f acc = accs[mi * 2 + ni];
            int col   = tn + ni * 16 + l16;
            int rbase = tm + mi * 16 + hi * 8;
            float bcol = bias ? bias[col] : 0.f;
#pragma unroll
            for (int r = 0; r < 8; ++r) {
                int row = rbase + r;
                float val = acc[r] + bcol;
                if (relu) val = fmaxf(val, 0.f);
                size_t o = (size_t)row * Nc + col;
                if (resid) val += resid[o];
                if (outF) outF[o] = val;
                else      outB[o] = __float2bfloat16(val);
            }
        }
    }
}

// ---------------------------------------------------------------------------
// KNN gather attention: one thread per (query n, head h). K=16, DH=32, H=8.
// logits = q.k/sqrt(32) + relu(pos@P1+pb1)@P2+pb2 ; softmax over K ; o = a.v
// ---------------------------------------------------------------------------
__global__ __launch_bounds__(256) void irn_attn(
    const __hip_bfloat16* __restrict__ q,
    const __hip_bfloat16* __restrict__ kf,
    const __hip_bfloat16* __restrict__ vf,
    const float* __restrict__ cq, const float* __restrict__ ckv,
    const int* __restrict__ idx,
    const float* __restrict__ P1, const float* __restrict__ pb1,
    const float* __restrict__ P2, const float* __restrict__ pb2,
    __hip_bfloat16* __restrict__ o, int Nq)
{
    int t = blockIdx.x * 256 + threadIdx.x;
    int n = t >> 3, h = t & 7;
    if (n >= Nq) return;

    float qv[32];
    const __hip_bfloat16* qp = q + (size_t)n * 256 + h * 32;
#pragma unroll
    for (int d = 0; d < 32; ++d) qv[d] = __bfloat162float(qp[d]);

    float c0 = cq[n * 3 + 0], c1 = cq[n * 3 + 1], c2 = cq[n * 3 + 2];
    float lg[16], mx = -1e30f;
    for (int j = 0; j < 16; ++j) {
        int ji = idx[n * 16 + j];
        float p0 = c0 - ckv[ji * 3 + 0];
        float p1 = c1 - ckv[ji * 3 + 1];
        float p2 = c2 - ckv[ji * 3 + 2];
        float bias = pb2[h];
#pragma unroll
        for (int p = 0; p < 32; ++p) {
            float tp = p0 * P1[p] + p1 * P1[32 + p] + p2 * P1[64 + p] + pb1[p];
            bias += fmaxf(tp, 0.f) * P2[p * 8 + h];
        }
        const __hip_bfloat16* kp = kf + (size_t)ji * 256 + h * 32;
        float dot = 0.f;
#pragma unroll
        for (int d = 0; d < 32; ++d) dot += qv[d] * __bfloat162float(kp[d]);
        float l = dot * 0.17677669529663687f + bias; // 1/sqrt(32)
        lg[j] = l;
        mx = fmaxf(mx, l);
    }
    float den = 0.f;
#pragma unroll
    for (int j = 0; j < 16; ++j) { lg[j] = expf(lg[j] - mx); den += lg[j]; }
    float inv = 1.f / den;

    float ov[32];
#pragma unroll
    for (int d = 0; d < 32; ++d) ov[d] = 0.f;
    for (int j = 0; j < 16; ++j) {
        int ji = idx[n * 16 + j];
        float w = lg[j] * inv;
        const __hip_bfloat16* vp = vf + (size_t)ji * 256 + h * 32;
#pragma unroll
        for (int d = 0; d < 32; ++d) ov[d] += w * __bfloat162float(vp[d]);
    }
    __hip_bfloat16* op = o + (size_t)n * 256 + h * 32;
#pragma unroll
    for (int d = 0; d < 32; ++d) op[d] = __float2bfloat16(ov[d]);
}

// ---------------------------------------------------------------------------
// Stage readout: sigmoid((featA @ Wkey + bkey) / 0.25), one wave per row
// ---------------------------------------------------------------------------
__global__ __launch_bounds__(256) void irn_readout(
    const float* __restrict__ x, const float* __restrict__ wk,
    const float* __restrict__ bk, float* __restrict__ out, int N)
{
    int wave = threadIdx.x >> 5, lane = threadIdx.x & 31;
    int row = blockIdx.x * 8 + wave;
    if (row >= N) return;
    const float* xr = x + (size_t)row * 256;
    float s = 0.f;
#pragma unroll
    for (int i = 0; i < 8; ++i) s += xr[lane + 32 * i] * wk[lane + 32 * i];
#pragma unroll
    for (int o = 16; o >= 1; o >>= 1) s += __shfl_xor(s, o, 32);
    if (lane == 0) {
        float logit = (s + bk[0]) * 4.0f; // / TEMP=0.25
        out[row] = 1.f / (1.f + expf(-logit));
    }
}

// ---------------------------------------------------------------------------
// Host orchestration
// ---------------------------------------------------------------------------
extern "C" void kernel_launch(void* const* d_in, const int* in_sizes, int n_in,
                              void* d_out, int out_size, void* d_ws, size_t ws_size,
                              hipStream_t stream)
{
    (void)in_sizes; (void)n_in; (void)out_size; (void)ws_size;
    const int NA = 32768, D = 256, DFF = 1024, Lc = 23, H = 8;

    const float* feat_a = (const float*)d_in[0];
    const float* feat_b = (const float*)d_in[1];
    const float* coord_a = (const float*)d_in[2];
    const float* coord_b = (const float*)d_in[3];
    const int* idxs[4] = { (const int*)d_in[4], (const int*)d_in[5],
                           (const int*)d_in[6], (const int*)d_in[7] };
    const float* Wq = (const float*)d_in[8];  const float* bq = (const float*)d_in[9];
    const float* Wk = (const float*)d_in[10]; const float* bk = (const float*)d_in[11];
    const float* Wv = (const float*)d_in[12]; const float* bv = (const float*)d_in[13];
    const float* Wo = (const float*)d_in[14]; const float* bo = (const float*)d_in[15];
    const float* ln1_g = (const float*)d_in[16]; const float* ln1_b = (const float*)d_in[17];
    const float* P1  = (const float*)d_in[18]; const float* pb1 = (const float*)d_in[19];
    const float* P2  = (const float*)d_in[20]; const float* pb2 = (const float*)d_in[21];
    const float* ln2_g = (const float*)d_in[22]; const float* ln2_b = (const float*)d_in[23];
    const float* W1 = (const float*)d_in[24]; const float* b1 = (const float*)d_in[25];
    const float* W2 = (const float*)d_in[26]; const float* b2 = (const float*)d_in[27];
    const float* Wkey = (const float*)d_in[28]; const float* bkey = (const float*)d_in[29];

    // ---- workspace carve-out ----
    char* wp = (char*)d_ws;
    auto take = [&](size_t bytes) -> void* {
        void* p = (void*)wp;
        wp += (bytes + 255) & ~(size_t)255;
        return p;
    };
    const size_t WPL = 4 * 65536 + 2 * 262144; // bf16 elems per layer (786432)
    __hip_bfloat16* wt   = (__hip_bfloat16*)take((size_t)Lc * WPL * 2);
    float* featA         = (float*)take((size_t)NA * D * 4);
    float* featB         = (float*)take((size_t)NA * D * 4);
    __hip_bfloat16* hq   = (__hip_bfloat16*)take((size_t)NA * D * 2);
    __hip_bfloat16* hkv  = (__hip_bfloat16*)take((size_t)NA * D * 2);
    __hip_bfloat16* qb   = (__hip_bfloat16*)take((size_t)NA * D * 2);
    __hip_bfloat16* kb   = (__hip_bfloat16*)take((size_t)NA * D * 2);
    __hip_bfloat16* vb   = (__hip_bfloat16*)take((size_t)NA * D * 2);
    __hip_bfloat16* f1   = (__hip_bfloat16*)take((size_t)NA * DFF * 2);

    hipMemcpyAsync(featA, feat_a, (size_t)NA * D * 4, hipMemcpyDeviceToDevice, stream);
    hipMemcpyAsync(featB, feat_b, (size_t)NA * D * 4, hipMemcpyDeviceToDevice, stream);

    // ---- weights: fp32 [K,N] -> bf16 [N,K] (stay resident in 192MB L2) ----
    const int gC256  = (256 * 256  + 255) / 256;
    const int gC1024 = (256 * 1024 + 255) / 256;
    for (int li = 0; li < Lc; ++li) {
        __hip_bfloat16* base = wt + (size_t)li * WPL;
        irn_convw<<<gC256,  256, 0, stream>>>(Wq + (size_t)li * 65536,  base,          256, 256);
        irn_convw<<<gC256,  256, 0, stream>>>(Wk + (size_t)li * 65536,  base + 65536,  256, 256);
        irn_convw<<<gC256,  256, 0, stream>>>(Wv + (size_t)li * 65536,  base + 131072, 256, 256);
        irn_convw<<<gC256,  256, 0, stream>>>(Wo + (size_t)li * 65536,  base + 196608, 256, 256);
        irn_convw<<<gC1024, 256, 0, stream>>>(W1 + (size_t)li * 262144, base + 262144, 256, 1024);
        irn_convw<<<gC1024, 256, 0, stream>>>(W2 + (size_t)li * 262144, base + 524288, 1024, 256);
    }

    // ---- schedule: 3 stages x 6 ops + 1 stage x 5 ops = 23 layers ----
    struct OpT { int qs, kv, ik; };
    const OpT ops[6] = { {0,1,2}, {1,0,3}, {0,0,0}, {1,1,1}, {0,1,2}, {1,0,3} };
    float* feats[2] = { featA, featB };
    const float* coords[2] = { coord_a, coord_b };

    const int gLN    = NA / 8;                        // 4096 blocks (8 rows/block)
    const int gG256  = (NA / 32) * (256 / 32) / 8;    // 1024 blocks (8 x 32x32 tiles)
    const int gG1024 = (NA / 32) * (1024 / 32) / 8;   // 4096 blocks
    const int gAttn  = NA * H / 256;                  // 1024 blocks

    int li = 0;
    for (int s = 0; s < 4; ++s) {
        int nops = (s < 3) ? 6 : 5;
        for (int oi = 0; oi < nops; ++oi) {
            int qs = ops[oi].qs, kv = ops[oi].kv;
            float* xq  = feats[qs];
            float* xkv = feats[kv];
            const int* idx = idxs[ops[oi].ik];
            __hip_bfloat16* base = wt + (size_t)li * WPL;

            // pre-norm
            irn_ln256<<<gLN, 256, 0, stream>>>(xq, ln1_g + li * 256, ln1_b + li * 256, hq, NA);
            __hip_bfloat16* hk = hq;
            if (qs != kv) {
                irn_ln256<<<gLN, 256, 0, stream>>>(xkv, ln1_g + li * 256, ln1_b + li * 256, hkv, NA);
                hk = hkv;
            }

            // Q/K/V projections (WMMA, bf16 out)
            irn_gemm_bf16<<<gG256, 256, 0, stream>>>(hq, base,          bq + li * 256, nullptr, nullptr, qb, NA, 256, 256, 0);
            irn_gemm_bf16<<<gG256, 256, 0, stream>>>(hk, base + 65536,  bk + li * 256, nullptr, nullptr, kb, NA, 256, 256, 0);
            irn_gemm_bf16<<<gG256, 256, 0, stream>>>(hk, base + 131072, bv + li * 256, nullptr, nullptr, vb, NA, 256, 256, 0);

            // KNN gather attention -> o (reuse hq buffer)
            irn_attn<<<gAttn, 256, 0, stream>>>(qb, kb, vb, coords[qs], coords[kv], idx,
                P1 + li * 96, pb1 + li * 32, P2 + li * 256, pb2 + li * 8, hq, NA);

            // O-projection + residual (fp32, in-place on feats)
            irn_gemm_bf16<<<gG256, 256, 0, stream>>>(hq, base + 196608, bo + li * 256, xq, xq, nullptr, NA, 256, 256, 0);

            // FFN: LN2 -> W1+ReLU (bf16) -> W2 + residual (fp32, in-place)
            irn_ln256<<<gLN, 256, 0, stream>>>(xq, ln2_g + li * 256, ln2_b + li * 256, hkv, NA);
            irn_gemm_bf16<<<gG1024, 256, 0, stream>>>(hkv, base + 262144, b1 + li * 1024, nullptr, nullptr, f1, NA, 256, 1024, 1);
            irn_gemm_bf16<<<gG256,  256, 0, stream>>>(f1,  base + 524288, b2 + li * 256,  xq, xq, nullptr, NA, 1024, 256, 0);
            ++li;
        }
        irn_readout<<<gLN, 256, 0, stream>>>(featA, Wkey + (size_t)s * 256, bkey + s,
                                             (float*)d_out + (size_t)s * NA, NA);
    }
}
